// CrossAttention_27522150433386
// MI455X (gfx1250) — compile-verified, matrix operand order checked by reading, not run on hardware
//
#include <hip/hip_runtime.h>

typedef __attribute__((ext_vector_type(16))) __bf16 bf16x16;
typedef __attribute__((ext_vector_type(8)))  float  floatx8;

#if defined(__has_builtin)
#if __has_builtin(__builtin_amdgcn_tensor_load_to_lds) && \
    __has_builtin(__builtin_amdgcn_s_wait_tensorcnt)
#define USE_TDM 1
#endif
#endif
#ifndef USE_TDM
#define USE_TDM 0
#endif

__device__ __forceinline__ unsigned short f32_to_bf16_rne(float f) {
  unsigned int u = __builtin_bit_cast(unsigned int, f);
  u += 0x7FFFu + ((u >> 16) & 1u);
  return (unsigned short)(u >> 16);
}

__device__ __forceinline__ floatx8 wmma_bf16(bf16x16 a, bf16x16 b, floatx8 c) {
  return __builtin_amdgcn_wmma_f32_16x16x32_bf16(false, a, false, b, (short)0, c,
                                                 false, false);
}

#if USE_TDM
typedef __attribute__((ext_vector_type(4))) unsigned int u32x4;
typedef __attribute__((ext_vector_type(8))) int i32x8;
typedef __attribute__((ext_vector_type(4))) int i32x4;

// Tensor Data Mover 2D tile load: tile_w contiguous bf16 per row, tile_h rows,
// row stride g_stride elements in global; packed tile_w*tile_h in LDS.
// D# layout per CDNA5 ISA ch.8: group0 = {count, lds_addr, global_addr, type=2},
// group1 = {data_size=2B, tensor dims == tile dims, dim0 stride}.
__device__ __forceinline__ void tdm_load_2d(unsigned short* lds_dst,
                                            const unsigned short* gsrc,
                                            unsigned tile_w, unsigned tile_h,
                                            unsigned g_stride) {
  unsigned long long ga = (unsigned long long)gsrc;
  unsigned la = (unsigned)(unsigned long long)(const void*)lds_dst;  // LDS byte offset
  u32x4 g0;
  g0[0] = 1u;                                                // count=1, user mode
  g0[1] = la;                                                // lds_addr
  g0[2] = (unsigned)ga;                                      // global_addr[31:0]
  g0[3] = (unsigned)((ga >> 32) & 0x01FFFFFFu) | (2u << 30); // addr[56:32] | type=2
  i32x8 g1;
  g1[0] = (int)(1u << 16);        // data_size code 1 = 2 bytes
  g1[1] = (int)(tile_w << 16);    // tensor_dim0[15:0] (bits 63:48)
  g1[2] = (int)(tile_h << 16);    // tensor_dim0 hi=0 | tensor_dim1[15:0]
  g1[3] = (int)(tile_w << 16);    // tensor_dim1 hi=0 | tile_dim0
  g1[4] = (int)tile_h;            // tile_dim1 | tile_dim2=0
  g1[5] = (int)g_stride;          // tensor_dim0_stride[31:0]
  g1[6] = 0;                      // stride0 hi | stride1 lo
  g1[7] = 0;
  i32x4 z4 = (i32x4)0;
  i32x8 z8 = (i32x8)0;
  // 6-arg form (clang-23 / therock-10.0 headers): groups 2,3 + extra group, cpol
  __builtin_amdgcn_tensor_load_to_lds(g0, g1, z4, z4, z8, 0);
}
__device__ __forceinline__ void tdm_wait() {
  __builtin_amdgcn_s_wait_tensorcnt(0);
}
#endif

// A-style 16x32 bf16 fragment from LDS [row][k], row stride ld.
__device__ __forceinline__ bf16x16 frag_rowmajor(const unsigned short* base,
                                                 int row, int ld, int k0) {
  const int half = (threadIdx.x >> 4) & 1;
  union { bf16x16 v; unsigned int u[8]; } f;
  const unsigned short* p = base + (size_t)row * ld + k0 + half * 8;
#pragma unroll
  for (int q = 0; q < 8; ++q) {
    int k = ((q & 3) << 1) + ((q >> 2) << 4);
    f.u[q] = *(const unsigned int*)(p + k);
  }
  return f.v;
}

// B-style 32x16 bf16 fragment from LDS [k][col], row stride ld.
__device__ __forceinline__ bf16x16 frag_colmajor(const unsigned short* base,
                                                 int col, int ld, int k0) {
  const int half = (threadIdx.x >> 4) & 1;
  union { bf16x16 v; unsigned short u[16]; } f;
  const unsigned short* p = base + col + (size_t)(k0 + half * 8) * ld;
#pragma unroll
  for (int e = 0; e < 16; ++e) {
    int q = e >> 1;
    int k = ((q & 3) << 1) + ((q >> 2) << 4) + (e & 1);
    f.u[e] = p[(size_t)k * ld];
  }
  return f.v;
}

__global__ __launch_bounds__(256) void cvt_f32_bf16_kernel(
    const float4* __restrict__ in, uint4* __restrict__ out, int n8) {
  int i = blockIdx.x * 256 + threadIdx.x;
  if (i >= n8) return;
  float4 a = in[2 * i], b = in[2 * i + 1];
  union { uint4 v; unsigned short u[8]; } o;
  o.u[0] = f32_to_bf16_rne(a.x); o.u[1] = f32_to_bf16_rne(a.y);
  o.u[2] = f32_to_bf16_rne(a.z); o.u[3] = f32_to_bf16_rne(a.w);
  o.u[4] = f32_to_bf16_rne(b.x); o.u[5] = f32_to_bf16_rne(b.y);
  o.u[6] = f32_to_bf16_rne(b.z); o.u[7] = f32_to_bf16_rne(b.w);
  out[i] = o.v;
}

// C[M,1024] = A[M,K] @ B[K,1024] + bias ; bf16 in, f32 accum.
// Block tile 128x128, 8 waves as 2(M)x4(N), wave tile 64x32. K staged via TDM
// (double buffered) or vector loads + prefetch.
template <bool OUTF>
__global__ __launch_bounds__(256) void gemm_bf16_kernel(
    const unsigned short* __restrict__ A, const unsigned short* __restrict__ Bw,
    const float* __restrict__ bias, unsigned short* __restrict__ Cb,
    float* __restrict__ Cf, int K) {
  constexpr int N = 1024;
  __shared__ __align__(16) unsigned short As[2][128 * 32];
  __shared__ __align__(16) unsigned short Bs[2][32 * 128];

  const int tid = threadIdx.x;
  const int lane15 = tid & 15;
  const int wm = (tid >> 5) >> 2;  // 0..1
  const int wn = (tid >> 5) & 3;   // 0..3
  const int m0 = blockIdx.y * 128;
  const int n0 = blockIdx.x * 128;

  floatx8 acc[4][2];
#pragma unroll
  for (int mt = 0; mt < 4; ++mt)
#pragma unroll
    for (int nt = 0; nt < 2; ++nt) acc[mt][nt] = (floatx8)(0.0f);

  const int nk = K >> 5;

#if USE_TDM
  if (tid < 32) {
    tdm_load_2d(As[0], A + (size_t)m0 * K, 32, 128, (unsigned)K);
    tdm_load_2d(Bs[0], Bw + n0, 128, 32, N);
    tdm_wait();
  }
  __syncthreads();
  for (int i = 0; i < nk; ++i) {
    const int cur = i & 1;
    if (tid < 32 && (i + 1) < nk) {  // prefetch next tile while computing
      tdm_load_2d(As[cur ^ 1], A + (size_t)m0 * K + (i + 1) * 32, 32, 128,
                  (unsigned)K);
      tdm_load_2d(Bs[cur ^ 1], Bw + (size_t)(i + 1) * 32 * N + n0, 128, 32, N);
    }
    bf16x16 af[4], bfr[2];
#pragma unroll
    for (int mt = 0; mt < 4; ++mt)
      af[mt] = frag_rowmajor(As[cur], wm * 64 + mt * 16 + lane15, 32, 0);
#pragma unroll
    for (int nt = 0; nt < 2; ++nt)
      bfr[nt] = frag_colmajor(Bs[cur], wn * 32 + nt * 16 + lane15, 128, 0);
#pragma unroll
    for (int mt = 0; mt < 4; ++mt)
#pragma unroll
      for (int nt = 0; nt < 2; ++nt)
        acc[mt][nt] = wmma_bf16(af[mt], bfr[nt], acc[mt][nt]);
    if (tid < 32) tdm_wait();
    __syncthreads();
  }
#else
  const int ar = tid >> 1, ac = (tid & 1) * 16;  // 128 x 32
  const int br = tid >> 3, bc = (tid & 7) * 16;  // 32 x 128
  for (int i = 0; i < nk; ++i) {
    const int k0 = i * 32;
    const uint4* ga = (const uint4*)(A + (size_t)(m0 + ar) * K + k0 + ac);
    uint4* sa = (uint4*)(As[0] + ar * 32 + ac);
    sa[0] = ga[0]; sa[1] = ga[1];
    const uint4* gb = (const uint4*)(Bw + (size_t)(k0 + br) * N + n0 + bc);
    uint4* sb = (uint4*)(Bs[0] + br * 128 + bc);
    sb[0] = gb[0]; sb[1] = gb[1];
    if (k0 + 32 < K) {
      __builtin_prefetch(A + (size_t)(m0 + ar) * K + k0 + 32 + ac, 0, 1);
      __builtin_prefetch(Bw + (size_t)(k0 + 32 + br) * N + n0 + bc, 0, 1);
    }
    __syncthreads();
    bf16x16 af[4], bfr[2];
#pragma unroll
    for (int mt = 0; mt < 4; ++mt)
      af[mt] = frag_rowmajor(As[0], wm * 64 + mt * 16 + lane15, 32, 0);
#pragma unroll
    for (int nt = 0; nt < 2; ++nt)
      bfr[nt] = frag_colmajor(Bs[0], wn * 32 + nt * 16 + lane15, 128, 0);
#pragma unroll
    for (int mt = 0; mt < 4; ++mt)
#pragma unroll
      for (int nt = 0; nt < 2; ++nt)
        acc[mt][nt] = wmma_bf16(af[mt], bfr[nt], acc[mt][nt]);
    __syncthreads();
  }
#endif

  const int half = (tid >> 4) & 1;
#pragma unroll
  for (int mt = 0; mt < 4; ++mt) {
#pragma unroll
    for (int nt = 0; nt < 2; ++nt) {
      const int colg = n0 + wn * 32 + nt * 16 + lane15;
      const int rowg = m0 + wm * 64 + mt * 16 + 8 * half;
      const float bval = bias[colg];
      if constexpr (OUTF) {
        float* p = Cf + (size_t)rowg * N + colg;
#pragma unroll
        for (int r = 0; r < 8; ++r) p[r * N] = acc[mt][nt][r] + bval;
      } else {
        unsigned short* p = Cb + (size_t)rowg * N + colg;
#pragma unroll
        for (int r = 0; r < 8; ++r)
          p[r * N] = f32_to_bf16_rne(acc[mt][nt][r] + bval);
      }
    }
  }
}

// Flash attention: block per (128-query tile, head, batch); 8 waves x 16 rows.
__global__ __launch_bounds__(256) void attn_kernel(
    const unsigned short* __restrict__ Q, const unsigned short* __restrict__ Kg,
    const unsigned short* __restrict__ Vg, unsigned short* __restrict__ O) {
  constexpr int E = 1024;
  __shared__ __align__(16) unsigned short Ks[128 * 64];
  __shared__ __align__(16) unsigned short Vs[128 * 64];
  __shared__ __align__(16) unsigned short QP[8 * 16 * 128];  // Q tile, then P

  const int tid = threadIdx.x;
  const int lane15 = tid & 15;
  const int half = (tid >> 4) & 1;
  const int w = tid >> 5;
  const int qt = blockIdx.x, h = blockIdx.y, b = blockIdx.z;
  const int rowQ0 = b * 2048 + qt * 128;
  const int rowK0 = b * 2048;
  const int colH = h * 64;

#if USE_TDM
  if (tid < 32) {
    tdm_load_2d(QP, Q + (size_t)rowQ0 * E + colH, 64, 128, E);
    tdm_wait();
  }
  __syncthreads();
#else
  for (int i = tid; i < (128 * 64) / 8; i += 256) {
    int lr = (i * 8) >> 6, d = (i * 8) & 63;
    *(uint4*)(QP + lr * 64 + d) =
        *(const uint4*)(Q + (size_t)(rowQ0 + lr) * E + colH + d);
  }
  __syncthreads();
#endif

  bf16x16 aQ[2];
#pragma unroll
  for (int t = 0; t < 2; ++t)
    aQ[t] = frag_rowmajor(QP, w * 16 + lane15, 64, t * 32);
  __syncthreads();  // QP now reused as per-wave P staging

  unsigned short* Pw = QP + w * 16 * 128;

  floatx8 Oacc[4];
#pragma unroll
  for (int dt = 0; dt < 4; ++dt) Oacc[dt] = (floatx8)(0.0f);
  float mrow[8], lrow[8];
#pragma unroll
  for (int r = 0; r < 8; ++r) { mrow[r] = -3.0e38f; lrow[r] = 0.0f; }

  for (int j = 0; j < 16; ++j) {
#if USE_TDM
    if (tid < 32) {
      tdm_load_2d(Ks, Kg + (size_t)(rowK0 + j * 128) * E + colH, 64, 128, E);
      tdm_load_2d(Vs, Vg + (size_t)(rowK0 + j * 128) * E + colH, 64, 128, E);
      tdm_wait();
    }
    __syncthreads();
#else
    for (int i = tid; i < (128 * 64) / 8; i += 256) {
      int lr = (i * 8) >> 6, d = (i * 8) & 63;
      *(uint4*)(Ks + lr * 64 + d) =
          *(const uint4*)(Kg + (size_t)(rowK0 + j * 128 + lr) * E + colH + d);
      *(uint4*)(Vs + lr * 64 + d) =
          *(const uint4*)(Vg + (size_t)(rowK0 + j * 128 + lr) * E + colH + d);
    }
    __syncthreads();
#endif

    // S = Q.K^T * scale (Ks[n][k] is row-major-by-n == B=K^T fragment layout)
    floatx8 sacc[8];
#pragma unroll
    for (int nt = 0; nt < 8; ++nt) {
      bf16x16 b0 = frag_rowmajor(Ks, nt * 16 + lane15, 64, 0);
      bf16x16 b1 = frag_rowmajor(Ks, nt * 16 + lane15, 64, 32);
      floatx8 c = wmma_bf16(aQ[0], b0, (floatx8)(0.0f));
      c = wmma_bf16(aQ[1], b1, c);
      sacc[nt] = c * 0.125f;  // 1/sqrt(64)
    }

    // online softmax; row = r + 8*half, cols across 16 lanes of the half
#pragma unroll
    for (int r = 0; r < 8; ++r) {
      float vmax = sacc[0][r];
#pragma unroll
      for (int nt = 1; nt < 8; ++nt) vmax = fmaxf(vmax, sacc[nt][r]);
#pragma unroll
      for (int off = 1; off < 16; off <<= 1)
        vmax = fmaxf(vmax, __shfl_xor(vmax, off, 16));
      float mNew = fmaxf(mrow[r], vmax);
      float alpha = __expf(mrow[r] - mNew);
      float rsum = 0.0f;
#pragma unroll
      for (int nt = 0; nt < 8; ++nt) {
        float p = __expf(sacc[nt][r] - mNew);
        sacc[nt][r] = p;
        rsum += p;
      }
#pragma unroll
      for (int off = 1; off < 16; off <<= 1) rsum += __shfl_xor(rsum, off, 16);
      mrow[r] = mNew;
      lrow[r] = lrow[r] * alpha + rsum;
#pragma unroll
      for (int dt = 0; dt < 4; ++dt) Oacc[dt][r] *= alpha;
    }

    // P (C-layout) -> bf16, transpose via per-wave LDS staging
#pragma unroll
    for (int nt = 0; nt < 8; ++nt)
#pragma unroll
      for (int r = 0; r < 8; ++r)
        Pw[(r + 8 * half) * 128 + nt * 16 + lane15] = f32_to_bf16_rne(sacc[nt][r]);
    __syncthreads();

    // O += P(16x128) x V(128x64)
#pragma unroll
    for (int t = 0; t < 4; ++t) {
      bf16x16 aP = frag_rowmajor(Pw, lane15, 128, t * 32);
#pragma unroll
      for (int dt = 0; dt < 4; ++dt) {
        bf16x16 bV = frag_colmajor(Vs, dt * 16 + lane15, 64, t * 32);
        Oacc[dt] = wmma_bf16(aP, bV, Oacc[dt]);
      }
    }
    __syncthreads();
  }

#pragma unroll
  for (int dt = 0; dt < 4; ++dt)
#pragma unroll
    for (int r = 0; r < 8; ++r) {
      int rowl = w * 16 + r + 8 * half;
      float v = Oacc[dt][r] / lrow[r];
      O[(size_t)(rowQ0 + rowl) * E + colH + dt * 16 + lane15] = f32_to_bf16_rne(v);
    }
}

extern "C" void kernel_launch(void* const* d_in, const int* in_sizes, int n_in,
                              void* d_out, int out_size, void* d_ws, size_t ws_size,
                              hipStream_t stream) {
  (void)in_sizes; (void)n_in; (void)out_size; (void)ws_size;
  const float* x   = (const float*)d_in[0];
  const float* ctx = (const float*)d_in[1];
  const float* Wq  = (const float*)d_in[2];
  const float* bq  = (const float*)d_in[3];
  const float* Wk  = (const float*)d_in[4];
  const float* bk  = (const float*)d_in[5];
  const float* Wv  = (const float*)d_in[6];
  const float* bv  = (const float*)d_in[7];
  const float* Wo  = (const float*)d_in[8];
  const float* bo  = (const float*)d_in[9];
  float* out = (float*)d_out;

  const int M = 4096, E = 1024, C = 768;  // M = B*S
  char* ws = (char*)d_ws;
  size_t off = 0;
  auto alloc_us = [&](size_t elems) -> unsigned short* {
    unsigned short* p = (unsigned short*)(ws + off);
    off += ((elems * 2) + 255) & ~(size_t)255;
    return p;
  };
  unsigned short* xb  = alloc_us((size_t)M * E);
  unsigned short* cb  = alloc_us((size_t)M * C);
  unsigned short* Wqb = alloc_us((size_t)E * E);
  unsigned short* Wkb = alloc_us((size_t)C * E);
  unsigned short* Wvb = alloc_us((size_t)C * E);
  unsigned short* Wob = alloc_us((size_t)E * E);
  unsigned short* Qb  = alloc_us((size_t)M * E);
  unsigned short* Kb  = alloc_us((size_t)M * E);
  unsigned short* Vb  = alloc_us((size_t)M * E);
  unsigned short* Ob  = alloc_us((size_t)M * E);

  auto cvt = [&](const float* src, unsigned short* dst, size_t n) {
    // all sizes divide 2048 = 256 threads * 8 elems
    cvt_f32_bf16_kernel<<<dim3((unsigned)(n / 2048)), 256, 0, stream>>>(
        (const float4*)src, (uint4*)dst, (int)(n / 8));
  };
  cvt(x,   xb,  (size_t)M * E);
  cvt(ctx, cb,  (size_t)M * C);
  cvt(Wq,  Wqb, (size_t)E * E);
  cvt(Wk,  Wkb, (size_t)C * E);
  cvt(Wv,  Wvb, (size_t)C * E);
  cvt(Wo,  Wob, (size_t)E * E);

  dim3 gg(E / 128, M / 128);
  gemm_bf16_kernel<false><<<gg, 256, 0, stream>>>(xb, Wqb, bq, Qb, nullptr, E);
  gemm_bf16_kernel<false><<<gg, 256, 0, stream>>>(cb, Wkb, bk, Kb, nullptr, C);
  gemm_bf16_kernel<false><<<gg, 256, 0, stream>>>(cb, Wvb, bv, Vb, nullptr, C);

  attn_kernel<<<dim3(16, 16, 2), 256, 0, stream>>>(Qb, Kb, Vb, Ob);

  gemm_bf16_kernel<true><<<gg, 256, 0, stream>>>(Ob, Wob, bo, nullptr, out, E);
}
